// MonolithGNN_32255204393241
// MI455X (gfx1250) — compile-verified, matrix operand order checked by reading, not run on hardware
//
#include <hip/hip_runtime.h>
#include <hip/hip_bf16.h>

// ---- problem constants (match reference) ----
#define NNODES 50000
#define NEDGES 400000
#define ETOT   (NEDGES + NNODES)   // self-loops appended
#define FH     256                 // H*HID
#define HEADS  4
#define HIDD   64
#define OUTC   16
#define NEG_SLOPE 0.2f

typedef __attribute__((ext_vector_type(16))) _Float16 v16h;
typedef __attribute__((ext_vector_type(8)))  _Float16 v8h;
typedef __attribute__((ext_vector_type(8)))  float    v8f;

#define GRID1(n) ((unsigned)(((long)(n) + 255) / 256))

// ---------- helpers: order-preserving float <-> u32 (for atomicMax segment-max) ----------
__device__ __forceinline__ unsigned enc_f32(float f) {
  unsigned u = __float_as_uint(f);
  return (u & 0x80000000u) ? ~u : (u | 0x80000000u);
}
__device__ __forceinline__ float dec_f32(unsigned k) {
  unsigned u = (k & 0x80000000u) ? (k & 0x7FFFFFFFu) : ~k;
  return __uint_as_float(u);
}

// ---------- elementwise utility kernels ----------
__global__ void k_zero_f32(float* p, long n) {
  long i = (long)blockIdx.x * blockDim.x + threadIdx.x;
  if (i < n) p[i] = 0.f;
}
__global__ void k_fill_u32(unsigned* p, long n, unsigned v) {
  long i = (long)blockIdx.x * blockDim.x + threadIdx.x;
  if (i < n) p[i] = v;
}
__global__ void k_cvt_f16(const float* __restrict__ s, _Float16* __restrict__ d, long n) {
  long i = (long)blockIdx.x * blockDim.x + threadIdx.x;
  if (i < n) d[i] = (_Float16)s[i];
}
// W: [K, Ncols] f32 row-major  ->  Wt: [Ncols, K] f16 (row n = column n of W)
__global__ void k_cvt_f16_t(const float* __restrict__ W, _Float16* __restrict__ Wt,
                            int K, int Ncols) {
  long i = (long)blockIdx.x * blockDim.x + threadIdx.x;
  if (i >= (long)K * Ncols) return;
  int nn = (int)(i / K);
  int kk = (int)(i % K);
  Wt[i] = (_Float16)W[(long)kk * Ncols + nn];
}
// X[i] = X[i] + b[i % cols]; act: 0=none, 1=ELU, 2=ReLU
__global__ void k_bias_act(float* __restrict__ X, const float* __restrict__ b,
                           long n, int cols, int act) {
  long i = (long)blockIdx.x * blockDim.x + threadIdx.x;
  if (i >= n) return;
  float v = X[i] + b[(int)(i % cols)];
  if (act == 1)      v = v > 0.f ? v : (__expf(v) - 1.f);
  else if (act == 2) v = v > 0.f ? v : 0.f;
  X[i] = v;
}

// ---------- WMMA GEMM: C[M,Ncols](f32) = A[M,K](f16,row-major) x Bt[Ncols,K](f16)^T ----------
// Block = 256 threads = 8 waves; each wave computes a 16 x (TN*16) strip of C.
//  * A fragment loaded once per k-step and reused across TN WMMAs (register reuse).
//  * The block's Bt slice [TN*16 rows x K] is staged in LDS once (contiguous copy,
//    global_load_b128 + ds_store_b128 + workgroup barrier), shared by all 8 waves.
// Fragment layout per ISA 7.12.2 (16-bit A 16x32): lanes 0-15 hold K 0-7 & 16-23,
// lanes 16-31 hold K 8-15 & 24-31 -> two contiguous 16B chunks per lane. B symmetric
// (column-per-lane) since Bt row = column of B. K mult of 32; M,Ncols mult of 16.
template<int TN>
__global__ void k_wmma_gemm(const _Float16* __restrict__ A, const _Float16* __restrict__ Bt,
                            float* __restrict__ C, int M, int Ncols, int K) {
  extern __shared__ _Float16 bs[];                    // [TN*16][K]
  const int ncg      = Ncols / (TN * 16);
  const int rowBlock = blockIdx.x / ncg;
  const int colGroup = blockIdx.x - rowBlock * ncg;

  { // cooperative stage of Bt slice (contiguous TN*16*K halves)
    const uint4* src = (const uint4*)(Bt + (size_t)colGroup * (TN * 16) * K);
    uint4*       dst = (uint4*)bs;
    const int   nvec = (TN * 16 * K) / 8;
    for (int i = threadIdx.x; i < nvec; i += blockDim.x) dst[i] = src[i];
  }
  __syncthreads();

  const int wave = threadIdx.x >> 5;
  const int lane = threadIdx.x & 31;
  const int tm   = rowBlock * 8 + wave;
  if (tm * 16 >= M) return;                           // wave-uniform -> EXEC all-ones
  const int r  = lane & 15;
  const int kg = (lane >> 4) << 3;                    // 0 or 8
  const _Float16* arow = A + (size_t)(tm * 16 + r) * K;

  v8f zero = {};
  v8f acc[TN];
#pragma unroll
  for (int t = 0; t < TN; t++) acc[t] = zero;

  for (int k0 = 0; k0 < K; k0 += 32) {
    union { v16h v; struct { v8h lo, hi; } p; } ua;
    ua.p.lo = *(const v8h*)(arow + k0 + kg);
    ua.p.hi = *(const v8h*)(arow + k0 + 16 + kg);
    if (k0 + 32 < K) __builtin_prefetch(arow + k0 + 32, 0, 1);   // global_prefetch_b8
#pragma unroll
    for (int t = 0; t < TN; t++) {
      union { v16h v; struct { v8h lo, hi; } p; } ub;
      const _Float16* brow = bs + (size_t)(t * 16 + r) * K;      // ds_load_b128 x2
      ub.p.lo = *(const v8h*)(brow + k0 + kg);
      ub.p.hi = *(const v8h*)(brow + k0 + 16 + kg);
      acc[t] = __builtin_amdgcn_wmma_f32_16x16x32_f16(false, ua.v, false, ub.v,
                                                      (short)0, acc[t], false, false);
    }
  }
  // C layout: VGPR v -> M = v + 8*(lane>=16), N = lane&15
  const int mo = (lane >> 4) << 3;
#pragma unroll
  for (int t = 0; t < TN; t++) {
    float* cbase = C + (size_t)(tm * 16) * Ncols + (colGroup * TN + t) * 16 + r;
#pragma unroll
    for (int v = 0; v < 8; v++) cbase[(size_t)(v + mo) * Ncols] = acc[t][v];
  }
}

// ---------- attention score precompute: es/ed[n,h] = <h[n,h,:], a_src/dst[h,:]> ----------
__global__ void k_attn_scores(const float* __restrict__ Hm, const float* __restrict__ asrc,
                              const float* __restrict__ adst, float* __restrict__ es,
                              float* __restrict__ ed, int n) {
  int i = blockIdx.x * blockDim.x + threadIdx.x;
  if (i >= n * HEADS) return;
  int node = i >> 2, h = i & 3;
  const float* row = Hm + (size_t)node * FH + h * HIDD;
  const float* as = asrc + h * HIDD;
  const float* ad = adst + h * HIDD;
  float s = 0.f, d = 0.f;
#pragma unroll 8
  for (int j = 0; j < HIDD; j++) { float v = row[j]; s += v * as[j]; d += v * ad[j]; }
  es[i] = s; ed[i] = d;
}

__device__ __forceinline__ void edge_sd(const int* __restrict__ ei, int e, int& s, int& d) {
  if (e < NEDGES) { s = ei[e]; d = ei[NEDGES + e]; }
  else            { s = e - NEDGES; d = s; }       // self-loop
}
__device__ __forceinline__ float edge_e(const float* es, const float* ed, int s, int d, int h) {
  float v = es[s * HEADS + h] + ed[d * HEADS + h];
  return v > 0.f ? v : NEG_SLOPE * v;              // leaky_relu
}

// pass 1: segment max via order-preserving u32 atomicMax
__global__ void k_edge_max(const int* __restrict__ ei, const float* __restrict__ es,
                           const float* __restrict__ ed, unsigned* __restrict__ mu) {
  long i = (long)blockIdx.x * blockDim.x + threadIdx.x;
  if (i >= (long)ETOT * HEADS) return;
  int e = (int)(i >> 2), h = (int)(i & 3);
  int s, d; edge_sd(ei, e, s, d);
  atomicMax(&mu[d * HEADS + h], enc_f32(edge_e(es, ed, s, d, h)));
}
// pass 2: segment sum of exp(e - m[dst])
__global__ void k_edge_sum(const int* __restrict__ ei, const float* __restrict__ es,
                           const float* __restrict__ ed, const unsigned* __restrict__ mu,
                           float* __restrict__ ssum) {
  long i = (long)blockIdx.x * blockDim.x + threadIdx.x;
  if (i >= (long)ETOT * HEADS) return;
  int e = (int)(i >> 2), h = (int)(i & 3);
  int s, d; edge_sd(ei, e, s, d);
  float p = __expf(edge_e(es, ed, s, d, h) - dec_f32(mu[d * HEADS + h]));
  atomicAdd(&ssum[d * HEADS + h], p);
}
// pass 3: out[dst] += h[src] * alpha  (one block per edge, 256 channels; L2-resident atomics)
__global__ void k_edge_agg(const int* __restrict__ ei, const float* __restrict__ Hm,
                           const float* __restrict__ es, const float* __restrict__ ed,
                           const unsigned* __restrict__ mu, const float* __restrict__ ssum,
                           float* __restrict__ outp) {
  int e = blockIdx.x;
  int t = threadIdx.x;          // channel 0..255
  int h = t >> 6;
  int s, d; edge_sd(ei, e, s, d);
  float p = __expf(edge_e(es, ed, s, d, h) - dec_f32(mu[d * HEADS + h]));
  float alpha = p / (ssum[d * HEADS + h] + 1e-16f);
  atomicAdd(&outp[(size_t)d * FH + t], Hm[(size_t)s * FH + t] * alpha);
}

// ---------- classifier epilogue: log_softmax(row + bias) over 16 classes ----------
__global__ void k_logsoftmax(const float* __restrict__ logits, const float* __restrict__ b,
                             float* __restrict__ out, int n) {
  int i = blockIdx.x * blockDim.x + threadIdx.x;
  if (i >= n) return;
  float v[OUTC]; float mx = -1e30f;
#pragma unroll
  for (int j = 0; j < OUTC; j++) { v[j] = logits[(size_t)i * OUTC + j] + b[j]; mx = fmaxf(mx, v[j]); }
  float s = 0.f;
#pragma unroll
  for (int j = 0; j < OUTC; j++) s += __expf(v[j] - mx);
  float l = logf(s);
#pragma unroll
  for (int j = 0; j < OUTC; j++) out[(size_t)i * OUTC + j] = v[j] - mx - l;
}

// =====================================================================================
extern "C" void kernel_launch(void* const* d_in, const int* in_sizes, int n_in,
                              void* d_out, int out_size, void* d_ws, size_t ws_size,
                              hipStream_t stream) {
  const float* x   = (const float*)d_in[0];
  const int*   ei  = (const int*)d_in[1];        // [2,E] int32 (JAX demotes int64)
  const float* W1  = (const float*)d_in[2];
  const float* a1s = (const float*)d_in[3];
  const float* a1d = (const float*)d_in[4];
  const float* b1  = (const float*)d_in[5];
  const float* W2  = (const float*)d_in[6];
  const float* a2s = (const float*)d_in[7];
  const float* a2d = (const float*)d_in[8];
  const float* b2  = (const float*)d_in[9];
  const float* W3  = (const float*)d_in[10];
  const float* a3s = (const float*)d_in[11];
  const float* a3d = (const float*)d_in[12];
  const float* b3  = (const float*)d_in[13];
  const float* pW1 = (const float*)d_in[14];
  const float* pb1 = (const float*)d_in[15];
  const float* pW2 = (const float*)d_in[16];
  const float* pb2 = (const float*)d_in[17];
  const float* cW1 = (const float*)d_in[18];
  const float* cb1 = (const float*)d_in[19];
  const float* cW2 = (const float*)d_in[20];
  const float* cb2 = (const float*)d_in[21];
  float* out = (float*)d_out;                    // [N*16 logits | N*64 proj]

  // ---- workspace carve-up (256B aligned) ----
  char* base = (char*)d_ws;
  size_t off = 0;
  auto take = [&](size_t bytes) { char* p = base + off; off = (off + bytes + 255) & ~(size_t)255; return p; };
  float*     bufA  = (float*)    take((size_t)NNODES * FH * 4);   // h (GEMM out)
  float*     bufB  = (float*)    take((size_t)NNODES * FH * 4);   // aggregation out / next input
  _Float16*  a16   = (_Float16*) take((size_t)NNODES * FH * 2);   // activations f16
  _Float16*  a16b  = (_Float16*) take((size_t)NNODES * HIDD * 2); // head activations f16
  _Float16*  w16t  = (_Float16*) take((size_t)FH * FH * 2);       // transposed weights f16
  float*     esb   = (float*)    take((size_t)NNODES * HEADS * 4);
  float*     edb   = (float*)    take((size_t)NNODES * HEADS * 4);
  unsigned*  mu    = (unsigned*) take((size_t)NNODES * HEADS * 4);
  float*     ssum  = (float*)    take((size_t)NNODES * HEADS * 4);
  float*     tcls  = (float*)    take((size_t)NNODES * HIDD * 4);
  float*     tproj = (float*)    take((size_t)NNODES * HIDD * 4);
  float*     lgts  = (float*)    take((size_t)NNODES * OUTC * 4);
  (void)ws_size; (void)n_in; (void)in_sizes; (void)out_size;

  const long nNF  = (long)NNODES * FH;
  const long nNH4 = (long)NNODES * HEADS;
  const long nE4  = (long)ETOT * HEADS;

  // grid for k_wmma_gemm<TN>: 8 row-tiles per block x (Ncols/(TN*16)) column groups
  auto gemm_grid = [](int M, int Ncols, int TN) {
    int mtiles = (M + 15) / 16;
    int rowBlocks = (mtiles + 7) / 8;
    return (unsigned)(rowBlocks * (Ncols / (TN * 16)));
  };
  auto gemm_shmem = [](int K, int TN) { return (size_t)TN * 16 * K * 2; };

  // ---- one GAT layer: X[N,FH] -> (bias+act)(segment_softmax_aggregate(X@W)) ----
  auto gat_layer = [&](const float* X, const float* W, const float* as_, const float* ad_,
                       const float* bias, float* Hm, float* Outm, int act) {
    k_cvt_f16  <<<GRID1(nNF), 256, 0, stream>>>(X, a16, nNF);
    k_cvt_f16_t<<<GRID1((long)FH * FH), 256, 0, stream>>>(W, w16t, FH, FH);
    k_wmma_gemm<4><<<gemm_grid(NNODES, FH, 4), 256, gemm_shmem(FH, 4), stream>>>(
        a16, w16t, Hm, NNODES, FH, FH);
    k_attn_scores<<<GRID1(nNH4), 256, 0, stream>>>(Hm, as_, ad_, esb, edb, NNODES);
    k_fill_u32 <<<GRID1(nNH4), 256, 0, stream>>>(mu, nNH4, 0u);   // enc(-inf) floor
    k_zero_f32 <<<GRID1(nNH4), 256, 0, stream>>>(ssum, nNH4);
    k_zero_f32 <<<GRID1(nNF), 256, 0, stream>>>(Outm, nNF);
    k_edge_max <<<GRID1(nE4), 256, 0, stream>>>(ei, esb, edb, mu);
    k_edge_sum <<<GRID1(nE4), 256, 0, stream>>>(ei, esb, edb, mu, ssum);
    k_edge_agg <<<ETOT, 256, 0, stream>>>(ei, Hm, esb, edb, mu, ssum, Outm);
    k_bias_act <<<GRID1(nNF), 256, 0, stream>>>(Outm, bias, nNF, FH, act);
  };

  gat_layer(x,    W1, a1s, a1d, b1, bufA, bufB, /*ELU*/1);
  gat_layer(bufB, W2, a2s, a2d, b2, bufA, bufB, /*ELU*/1);
  gat_layer(bufB, W3, a3s, a3d, b3, bufA, bufB, /*none*/0);   // h_final = bufB

  // ---- heads ----
  k_cvt_f16<<<GRID1(nNF), 256, 0, stream>>>(bufB, a16, nNF);

  // classifier stage 1: relu(h @ cW1 + cb1)
  k_cvt_f16_t<<<GRID1((long)FH * HIDD), 256, 0, stream>>>(cW1, w16t, FH, HIDD);
  k_wmma_gemm<4><<<gemm_grid(NNODES, HIDD, 4), 256, gemm_shmem(FH, 4), stream>>>(
      a16, w16t, tcls, NNODES, HIDD, FH);
  k_bias_act <<<GRID1((long)NNODES * HIDD), 256, 0, stream>>>(tcls, cb1, (long)NNODES * HIDD, HIDD, 2);

  // projection stage 1: relu(h @ pW1 + pb1)
  k_cvt_f16_t<<<GRID1((long)FH * HIDD), 256, 0, stream>>>(pW1, w16t, FH, HIDD);
  k_wmma_gemm<4><<<gemm_grid(NNODES, HIDD, 4), 256, gemm_shmem(FH, 4), stream>>>(
      a16, w16t, tproj, NNODES, HIDD, FH);
  k_bias_act <<<GRID1((long)NNODES * HIDD), 256, 0, stream>>>(tproj, pb1, (long)NNODES * HIDD, HIDD, 2);

  // classifier stage 2: log_softmax(t_cls @ cW2 + cb2) -> out[0 : N*16)
  k_cvt_f16  <<<GRID1((long)NNODES * HIDD), 256, 0, stream>>>(tcls, a16b, (long)NNODES * HIDD);
  k_cvt_f16_t<<<GRID1((long)HIDD * OUTC), 256, 0, stream>>>(cW2, w16t, HIDD, OUTC);
  k_wmma_gemm<1><<<gemm_grid(NNODES, OUTC, 1), 256, gemm_shmem(HIDD, 1), stream>>>(
      a16b, w16t, lgts, NNODES, OUTC, HIDD);
  k_logsoftmax<<<GRID1(NNODES), 256, 0, stream>>>(lgts, cb2, out, NNODES);

  // projection stage 2: t_proj @ pW2 + pb2 -> out[N*16 : N*16 + N*64)
  float* proj_out = out + (size_t)NNODES * OUTC;
  k_cvt_f16  <<<GRID1((long)NNODES * HIDD), 256, 0, stream>>>(tproj, a16b, (long)NNODES * HIDD);
  k_cvt_f16_t<<<GRID1((long)HIDD * HIDD), 256, 0, stream>>>(pW2, w16t, HIDD, HIDD);
  k_wmma_gemm<4><<<gemm_grid(NNODES, HIDD, 4), 256, gemm_shmem(HIDD, 4), stream>>>(
      a16b, w16t, proj_out, NNODES, HIDD, HIDD);
  k_bias_act <<<GRID1((long)NNODES * HIDD), 256, 0, stream>>>(proj_out, pb2, (long)NNODES * HIDD, HIDD, 0);
}
